// OptimizedMOE_67242007986687
// MI455X (gfx1250) — matrix operand order repaired and checked by reading, not compiled, not run on hardware
//
#include <hip/hip_runtime.h>
#include <hip/hip_bf16.h>

typedef __bf16 bf16;
typedef __attribute__((ext_vector_type(16))) __bf16 bf16x16;
typedef __attribute__((ext_vector_type(8)))  __bf16 bf16x8;
typedef __attribute__((ext_vector_type(8)))  float  v8f;
typedef __attribute__((ext_vector_type(4)))  float  v4f;
typedef __attribute__((ext_vector_type(4)))  unsigned int u32x4;
typedef __attribute__((ext_vector_type(8)))  int  i32x8;
typedef __attribute__((ext_vector_type(4)))  int  i32x4;

#define HIDDEN   1024
#define NEXP     16
#define DFF_E    512
#define DFF_S    2048
#define TOKENS   4096

#define BM       64
#define BN       128
#define BK       32
#define NTHREADS 256

#define WMMA_BF16(a, b, c) \
  __builtin_amdgcn_wmma_f32_16x16x32_bf16(false, (a), false, (b), (short)0, (c), false, false)

// ---------------------------------------------------------------------------
// Tensor Data Mover: 2D bf16 tile -> LDS  (D# per CDNA5 ISA)
// Source is row-major [rows][row_len]; tile = tile_d1 rows x tile_d0 elems,
// stride_d0 elems between rows.  Lands contiguous [tile_d1][tile_d0] in LDS,
// which is exactly the frag_b / frag_a LDS layout (pitch BK).
// ---------------------------------------------------------------------------
__device__ __forceinline__ void tdm_load_2d_bf16(unsigned lds_byte_addr,
                                                 const bf16* gptr,
                                                 unsigned tensor_d0, unsigned tensor_d1,
                                                 unsigned tile_d0,   unsigned tile_d1,
                                                 unsigned stride_d0) {
  unsigned long long ga = (unsigned long long)(size_t)gptr;
  u32x4 g0;
  g0[0] = 1u;                                                 // count=1, user D#
  g0[1] = lds_byte_addr;                                      // lds_addr[63:32]
  g0[2] = (unsigned)(ga & 0xFFFFFFFFu);                       // global_addr[95:64]
  g0[3] = (unsigned)((ga >> 32) & 0x01FFFFFFu) | (2u << 30);  // ga[120:96], type=2
  i32x8 g1;
  g1[0] = (int)(1u << 16);                                    // data_size=1 (2B)
  g1[1] = (int)((tensor_d0 & 0xFFFFu) << 16);                 // tensor_dim0[15:0]
  g1[2] = (int)((tensor_d0 >> 16) | ((tensor_d1 & 0xFFFFu) << 16));
  g1[3] = (int)((tensor_d1 >> 16) | (tile_d0 << 16));         // tile_dim0
  g1[4] = (int)(tile_d1 & 0xFFFFu);                           // tile_dim1, tile_dim2=0
  g1[5] = (int)stride_d0;                                     // dim0_stride[31:0]
  g1[6] = 0;
  g1[7] = 0;
  i32x4 z4 = {};
  i32x8 z8 = {};                                              // groups 2/3 unused (2D)
  __builtin_amdgcn_tensor_load_to_lds(g0, g1, z4, z4, z8, 0);
}

__device__ __forceinline__ unsigned lds_addr_of(const void* p) {
  return (unsigned)(size_t)p;   // LDS aperture: addr[31:0] is the LDS offset
}

// ---------------------------------------------------------------------------
// Fragment loads (CDNA5 16-bit WMMA VGPR striping)
// ---------------------------------------------------------------------------
__device__ __forceinline__ bf16x16 frag_a(const bf16* As, int mbase, int lane) {
  const bf16* p = As + (mbase + (lane & 15)) * BK + ((lane >> 4) * 16);
  bf16x8 lo = *(const bf16x8*)p;
  bf16x8 hi = *(const bf16x8*)(p + 8);
  bf16x16 r;
  #pragma unroll
  for (int i = 0; i < 8; ++i) { r[i] = lo[i]; r[i + 8] = hi[i]; }
  return r;
}

__device__ __forceinline__ bf16x16 frag_b(const bf16* Bs, int nbase, int lane) {
  const bf16* p = Bs + (nbase + (lane & 15)) * BK + ((lane >> 4) * 16);
  return *(const bf16x16*)p;
}

__device__ __forceinline__ float silu_f(float g) {
  return g / (1.0f + __expf(-g));
}

// ---------------------------------------------------------------------------
// Kernel 0: zero expert counters
// ---------------------------------------------------------------------------
__global__ void moe_zero_kernel(int* __restrict__ cnts) {
  if (threadIdx.x < NEXP) cnts[threadIdx.x] = 0;
}

// ---------------------------------------------------------------------------
// Kernel 1: fp32 -> bf16 copy (activations)
// ---------------------------------------------------------------------------
__global__ __launch_bounds__(256)
void moe_cvt_kernel(const float* __restrict__ in, bf16* __restrict__ out, long n) {
  long i = ((long)blockIdx.x * 256 + threadIdx.x) * 8;
  if (i >= n) return;
  v4f a = *(const v4f*)(in + i);
  v4f b = *(const v4f*)(in + i + 4);
  bf16x8 o;
  #pragma unroll
  for (int j = 0; j < 4; ++j) { o[j] = (bf16)a[j]; o[4 + j] = (bf16)b[j]; }
  *(bf16x8*)(out + i) = o;
}

// ---------------------------------------------------------------------------
// Kernel 2: fp32 [K][N] -> bf16 transposed [N][K]   (per-expert via blockIdx.z)
// ---------------------------------------------------------------------------
__global__ __launch_bounds__(256)
void moe_cvt_t_kernel(const float* __restrict__ in, bf16* __restrict__ out,
                      int K, int N) {
  int e = blockIdx.z;
  const float* src = in  + (size_t)e * K * N;
  bf16*        dst = out + (size_t)e * N * K;
  int k0 = blockIdx.x * 64, n0 = blockIdx.y * 64;
  __shared__ __align__(64) bf16 t[64][64];
  int tid = threadIdx.x;
  int kr = tid >> 2;
  int nc = (tid & 3) * 16;
  const float* s = src + (size_t)(k0 + kr) * N + n0 + nc;
  #pragma unroll
  for (int j = 0; j < 16; ++j) t[nc + j][kr] = (bf16)s[j];
  __syncthreads();
  int nr = tid >> 2;
  int kc = (tid & 3) * 16;
  *(bf16x16*)(dst + (size_t)(n0 + nr) * K + k0 + kc) = *(const bf16x16*)&t[nr][kc];
}

// ---------------------------------------------------------------------------
// Kernel 3: router — logits, softmax, top-2, gathered expert lists
// ---------------------------------------------------------------------------
__global__ __launch_bounds__(256)
void moe_router_kernel(const float* __restrict__ X, const float* __restrict__ RW,
                       int* __restrict__ cnts, int* __restrict__ lists,
                       float* __restrict__ tw) {
  int t = blockIdx.x * blockDim.x + threadIdx.x;
  if (t >= TOKENS) return;
  float acc[NEXP];
  #pragma unroll
  for (int e = 0; e < NEXP; ++e) acc[e] = 0.0f;
  const float* xr = X + (size_t)t * HIDDEN;
  for (int k = 0; k < HIDDEN; ++k) {
    float xv = xr[k];
    const float* w = RW + (size_t)k * NEXP;
    #pragma unroll
    for (int e = 0; e < NEXP; ++e) acc[e] = fmaf(xv, w[e], acc[e]);
  }
  float mx = acc[0];
  #pragma unroll
  for (int e = 1; e < NEXP; ++e) mx = fmaxf(mx, acc[e]);
  float sum = 0.0f;
  #pragma unroll
  for (int e = 0; e < NEXP; ++e) { acc[e] = __expf(acc[e] - mx); sum += acc[e]; }
  float inv = 1.0f / sum;
  int e0 = 0, e1 = -1;
  float b0 = -1.0f, b1 = -1.0f;
  #pragma unroll
  for (int e = 0; e < NEXP; ++e) {
    float v = acc[e];
    if (v > b0)      { b1 = b0; e1 = e0; b0 = v; e0 = e; }
    else if (v > b1) { b1 = v; e1 = e; }
  }
  tw[2 * t]     = b0 * inv;
  tw[2 * t + 1] = b1 * inv;
  int p0 = atomicAdd(&cnts[e0], 1);
  lists[e0 * TOKENS + p0] = 2 * t;
  int p1 = atomicAdd(&cnts[e1], 1);
  lists[e1 * TOKENS + p1] = 2 * t + 1;
}

// ---------------------------------------------------------------------------
// Kernel 4: fused gate+up GEMM + SwiGLU.  A: bf16 (gathered for experts,
// gather hoisted out of the K loop), B: bf16 [N][K] weights streamed by TDM,
// double-buffered.  Shared: out row = token.  Expert: out row = list entry.
// ---------------------------------------------------------------------------
__global__ __launch_bounds__(NTHREADS)
void moe_gateup_kernel(const bf16* __restrict__ X,
                       const bf16* __restrict__ WgT,
                       const bf16* __restrict__ WuT,
                       bf16* __restrict__ Out,
                       int K, int N,
                       const int* __restrict__ lists,
                       const int* __restrict__ cnts) {
  int e  = blockIdx.z;
  int m0 = blockIdx.y * BM;
  int n0 = blockIdx.x * BN;
  int cnt = TOKENS;
  const int* gl = nullptr;
  if (cnts) {
    cnt = cnts[e];
    if (m0 >= cnt) return;
    gl = lists + (size_t)e * TOKENS;
  }
  const bf16* wg = WgT + (size_t)e * K * N;   // [N][K], e==0 for shared
  const bf16* wu = WuT + (size_t)e * K * N;

  __shared__ __align__(64) bf16 As [BM * BK];
  __shared__ __align__(64) bf16 BsG[2][BN * BK];
  __shared__ __align__(64) bf16 BsU[2][BN * BK];

  int tid  = threadIdx.x;
  int lane = tid & 31;
  int wave = tid >> 5;
  int mW   = (wave & 1) * 32;
  int nW   = (wave >> 1) * 32;

  // hoisted A-stage source pointer (gather is K-loop invariant)
  int arow = tid >> 2;               // 0..63
  int akc  = (tid & 3) * 8;          // 0,8,16,24
  bool avalid = (m0 + arow) < cnt;
  const bf16* asrc = nullptr;
  if (avalid) {
    int sr = gl ? (gl[m0 + arow] >> 1) : (m0 + arow);
    asrc = X + (size_t)sr * K + akc;
  }
  bf16x8* adst = (bf16x8*)(As + arow * BK + akc);

  v8f accG[2][2] = {};
  v8f accU[2][2] = {};

  int iters = K / BK;
  if (wave == 0) {   // prologue: DMA first B tiles into buffer 0
    tdm_load_2d_bf16(lds_addr_of(&BsG[0][0]), wg + (size_t)n0 * K,
                     (unsigned)K, (unsigned)BN, BK, BN, (unsigned)K);
    tdm_load_2d_bf16(lds_addr_of(&BsU[0][0]), wu + (size_t)n0 * K,
                     (unsigned)K, (unsigned)BN, BK, BN, (unsigned)K);
  }

  for (int it = 0; it < iters; ++it) {
    int k0 = it * BK;
    __syncthreads();                       // prev compute done; buffers reusable
    if (avalid) *adst = *(const bf16x8*)(asrc + k0);
    else        { bf16x8 zz = {}; *adst = zz; }
    if (wave == 0) {
      if (it + 1 < iters) {
        int buf = (it + 1) & 1;
        tdm_load_2d_bf16(lds_addr_of(&BsG[buf][0]), wg + (size_t)n0 * K + (k0 + BK),
                         (unsigned)K, (unsigned)BN, BK, BN, (unsigned)K);
        tdm_load_2d_bf16(lds_addr_of(&BsU[buf][0]), wu + (size_t)n0 * K + (k0 + BK),
                         (unsigned)K, (unsigned)BN, BK, BN, (unsigned)K);
        __builtin_amdgcn_s_wait_tensorcnt((short)2);   // current tiles landed
      } else {
        __builtin_amdgcn_s_wait_tensorcnt((short)0);
      }
    }
    __syncthreads();                       // A + current B tiles visible

    int buf = it & 1;
    bf16x16 a0 = frag_a(As, mW,      lane);
    bf16x16 a1 = frag_a(As, mW + 16, lane);
    #pragma unroll
    for (int tn = 0; tn < 2; ++tn) {
      bf16x16 bg = frag_b(&BsG[buf][0], nW + tn * 16, lane);
      bf16x16 bu = frag_b(&BsU[buf][0], nW + tn * 16, lane);
      accG[0][tn] = WMMA_BF16(a0, bg, accG[0][tn]);
      accG[1][tn] = WMMA_BF16(a1, bg, accG[1][tn]);
      accU[0][tn] = WMMA_BF16(a0, bu, accU[0][tn]);
      accU[1][tn] = WMMA_BF16(a1, bu, accU[1][tn]);
    }
  }

  int col  = lane & 15;
  int rofs = (lane >> 4) * 8;
  #pragma unroll
  for (int tm = 0; tm < 2; ++tm) {
    #pragma unroll
    for (int tn = 0; tn < 2; ++tn) {
      #pragma unroll
      for (int r = 0; r < 8; ++r) {
        int p = m0 + mW + tm * 16 + rofs + r;
        if (p < cnt) {
          int orow = gl ? gl[p] : p;       // entry-indexed for experts
          float g = accG[tm][tn][r];
          float u = accU[tm][tn][r];
          int n = n0 + nW + tn * 16 + col;
          Out[(size_t)orow * N + n] = (bf16)(silu_f(g) * u);
        }
      }
    }
  }
}

// ---------------------------------------------------------------------------
// Kernel 5: down-projection GEMM.  B streamed by TDM (double-buffered).
// Shared: plain store to out.  Expert: A rows entry-indexed (hoisted),
// weighted atomicAdd scatter into out.
// ---------------------------------------------------------------------------
__global__ __launch_bounds__(NTHREADS)
void moe_down_kernel(const bf16* __restrict__ Ain,
                     const bf16* __restrict__ WdT,   // [HIDDEN][K] bf16
                     float* __restrict__ Out,
                     int K,
                     const int* __restrict__ lists,
                     const int* __restrict__ cnts,
                     const float* __restrict__ tw) {
  int e  = blockIdx.z;
  int m0 = blockIdx.y * BM;
  int n0 = blockIdx.x * BN;
  int cnt = TOKENS;
  const int* gl = nullptr;
  if (cnts) {
    cnt = cnts[e];
    if (m0 >= cnt) return;
    gl = lists + (size_t)e * TOKENS;
  }
  const bf16* wd = WdT + (size_t)e * K * HIDDEN;

  __shared__ __align__(64) bf16 As[BM * BK];
  __shared__ __align__(64) bf16 Bs[2][BN * BK];

  int tid  = threadIdx.x;
  int lane = tid & 31;
  int wave = tid >> 5;
  int mW   = (wave & 1) * 32;
  int nW   = (wave >> 1) * 32;

  int arow = tid >> 2;
  int akc  = (tid & 3) * 8;
  bool avalid = (m0 + arow) < cnt;
  const bf16* asrc = nullptr;
  if (avalid) {
    int sr = gl ? gl[m0 + arow] : (m0 + arow);   // entry-indexed rows
    asrc = Ain + (size_t)sr * K + akc;
  }
  bf16x8* adst = (bf16x8*)(As + arow * BK + akc);

  v8f acc[2][2] = {};

  int iters = K / BK;
  if (wave == 0) {
    tdm_load_2d_bf16(lds_addr_of(&Bs[0][0]), wd + (size_t)n0 * K,
                     (unsigned)K, (unsigned)BN, BK, BN, (unsigned)K);
  }

  for (int it = 0; it < iters; ++it) {
    int k0 = it * BK;
    __syncthreads();
    if (avalid) *adst = *(const bf16x8*)(asrc + k0);
    else        { bf16x8 zz = {}; *adst = zz; }
    if (wave == 0) {
      if (it + 1 < iters) {
        int buf = (it + 1) & 1;
        tdm_load_2d_bf16(lds_addr_of(&Bs[buf][0]), wd + (size_t)n0 * K + (k0 + BK),
                         (unsigned)K, (unsigned)BN, BK, BN, (unsigned)K);
        __builtin_amdgcn_s_wait_tensorcnt((short)1);
      } else {
        __builtin_amdgcn_s_wait_tensorcnt((short)0);
      }
    }
    __syncthreads();

    int buf = it & 1;
    bf16x16 a0 = frag_a(As, mW,      lane);
    bf16x16 a1 = frag_a(As, mW + 16, lane);
    #pragma unroll
    for (int tn = 0; tn < 2; ++tn) {
      bf16x16 b = frag_b(&Bs[buf][0], nW + tn * 16, lane);
      acc[0][tn] = WMMA_BF16(a0, b, acc[0][tn]);
      acc[1][tn] = WMMA_BF16(a1, b, acc[1][tn]);
    }
  }

  int col  = lane & 15;
  int rofs = (lane >> 4) * 8;
  #pragma unroll
  for (int tm = 0; tm < 2; ++tm) {
    #pragma unroll
    for (int tn = 0; tn < 2; ++tn) {
      #pragma unroll
      for (int r = 0; r < 8; ++r) {
        int p = m0 + mW + tm * 16 + rofs + r;
        if (p < cnt) {
          int n = n0 + nW + tn * 16 + col;
          float v = acc[tm][tn][r];
          if (cnts) {
            int entry = gl[p];
            atomicAdd(Out + (size_t)(entry >> 1) * HIDDEN + n, tw[entry] * v);
          } else {
            Out[(size_t)p * HIDDEN + n] = v;
          }
        }
      }
    }
  }
}

// ---------------------------------------------------------------------------
// Launch
// ---------------------------------------------------------------------------
extern "C" void kernel_launch(void* const* d_in, const int* in_sizes, int n_in,
                              void* d_out, int out_size, void* d_ws, size_t ws_size,
                              hipStream_t stream) {
  (void)in_sizes; (void)n_in; (void)out_size; (void)ws_size;
  const float* x  = (const float*)d_in[0];
  const float* rw = (const float*)d_in[1];
  const float* eg = (const float*)d_in[2];
  const float* eu = (const float*)d_in[3];
  const float* ed = (const float*)d_in[4];
  const float* sg = (const float*)d_in[5];
  const float* su = (const float*)d_in[6];
  const float* sd = (const float*)d_in[7];
  float* out = (float*)d_out;

  // workspace carve-up (~92 MB)
  char* w = (char*)d_ws;
  int*   cnts  = (int*)w;    w += 256;
  int*   lists = (int*)w;    w += (size_t)NEXP * TOKENS * sizeof(int);
  float* tw    = (float*)w;  w += (size_t)TOKENS * 2 * sizeof(float);
  bf16* xb     = (bf16*)w;   w += (size_t)TOKENS * HIDDEN * sizeof(bf16);
  bf16* sgT    = (bf16*)w;   w += (size_t)DFF_S * HIDDEN * sizeof(bf16);
  bf16* suT    = (bf16*)w;   w += (size_t)DFF_S * HIDDEN * sizeof(bf16);
  bf16* sdT    = (bf16*)w;   w += (size_t)HIDDEN * DFF_S * sizeof(bf16);
  bf16* egT    = (bf16*)w;   w += (size_t)NEXP * DFF_E * HIDDEN * sizeof(bf16);
  bf16* euT    = (bf16*)w;   w += (size_t)NEXP * DFF_E * HIDDEN * sizeof(bf16);
  bf16* edT    = (bf16*)w;   w += (size_t)NEXP * HIDDEN * DFF_E * sizeof(bf16);
  bf16* interS = (bf16*)w;   w += (size_t)TOKENS * DFF_S * sizeof(bf16);
  bf16* interE = (bf16*)w;   // [2*TOKENS][DFF_E], entry-indexed

  moe_zero_kernel<<<1, 32, 0, stream>>>(cnts);
  moe_router_kernel<<<TOKENS / 256, 256, 0, stream>>>(x, rw, cnts, lists, tw);

  // activations fp32 -> bf16
  {
    long n = (long)TOKENS * HIDDEN;
    moe_cvt_kernel<<<(unsigned)(n / 8 / 256), 256, 0, stream>>>(x, xb, n);
  }
  // weights fp32 [K][N] -> bf16 [N][K]
  moe_cvt_t_kernel<<<dim3(HIDDEN / 64, DFF_S / 64, 1), 256, 0, stream>>>(sg, sgT, HIDDEN, DFF_S);
  moe_cvt_t_kernel<<<dim3(HIDDEN / 64, DFF_S / 64, 1), 256, 0, stream>>>(su, suT, HIDDEN, DFF_S);
  moe_cvt_t_kernel<<<dim3(DFF_S / 64, HIDDEN / 64, 1), 256, 0, stream>>>(sd, sdT, DFF_S, HIDDEN);
  moe_cvt_t_kernel<<<dim3(HIDDEN / 64, DFF_E / 64, NEXP), 256, 0, stream>>>(eg, egT, HIDDEN, DFF_E);
  moe_cvt_t_kernel<<<dim3(HIDDEN / 64, DFF_E / 64, NEXP), 256, 0, stream>>>(eu, euT, HIDDEN, DFF_E);
  moe_cvt_t_kernel<<<dim3(DFF_E / 64, HIDDEN / 64, NEXP), 256, 0, stream>>>(ed, edT, DFF_E, HIDDEN);

  // shared SwiGLU
  moe_gateup_kernel<<<dim3(DFF_S / BN, TOKENS / BM, 1), NTHREADS, 0, stream>>>(
      xb, sgT, suT, interS, HIDDEN, DFF_S, nullptr, nullptr);
  moe_down_kernel<<<dim3(HIDDEN / BN, TOKENS / BM, 1), NTHREADS, 0, stream>>>(
      interS, sdT, out, DFF_S, nullptr, nullptr, nullptr);

  // routed experts (gathered)
  moe_gateup_kernel<<<dim3(DFF_E / BN, TOKENS / BM, NEXP), NTHREADS, 0, stream>>>(
      xb, egT, euT, interE, HIDDEN, DFF_E, lists, cnts);
  moe_down_kernel<<<dim3(HIDDEN / BN, TOKENS / BM, NEXP), NTHREADS, 0, stream>>>(
      interE, edT, out, DFF_E, lists, cnts, tw);
}
